// EnhancedTokenwiseAggregation_8693013807679
// MI455X (gfx1250) — matrix operand order repaired and checked by reading, not compiled, not run on hardware
//
#include <hip/hip_runtime.h>
#include <cstdint>

// ---------------------------------------------------------------------------
// EnhancedTokenwiseAggregation — single-query multi-head attention aggregation
// Shapes (fixed by reference): S=4, B=4, L=4096, E=1024, H=8, D=128
//
// Math refactor:
//   qk[h,e] = scale * sum_d q[h*D+d] * Wk[h*D+d, e]   (kills the K projection)
//   logits[t,s,h] = qk[h,:] . x[t,s,:] + qb[h]
//   attn = softmax_s(logits)
//   y[t,h,:] = sum_s attn[t,h,s] * x[t,s,:]
//   ctx[t,h,:] = Wv_h @ y[t,h,:] + bv_h               (8 head-blocked GEMMs)
//   agg[t,:]   = Wout @ ctx[t,:] + bout               (one big GEMM)
// Heavy GEMMs run on v_wmma_f32_16x16x32_bf16; the x-tile staging uses
// global_load_async_to_lds_b128 with double buffering (ASYNCcnt-tracked).
// ---------------------------------------------------------------------------

constexpr int kS = 4;
constexpr int kB = 4;
constexpr int kL = 4096;
constexpr int kE = 1024;
constexpr int kH = 8;
constexpr int kD = 128;
constexpr int kNTok = kB * kL;            // 16384 tokens
constexpr float kScale = 0.08838834764831845f;  // 1/sqrt(128)

typedef __bf16 v16bf __attribute__((ext_vector_type(16)));
typedef __bf16 v8bf  __attribute__((ext_vector_type(8)));
typedef float  v8f   __attribute__((ext_vector_type(8)));
typedef float  v4f   __attribute__((ext_vector_type(4)));

// Workspace layout (bytes). Assumes ws_size >= ~38 MB.
constexpr size_t WS_QK   = 0;                          // f32 [H][E]      32 KB
constexpr size_t WS_QB   = 32768;                      // f32 [H]
constexpr size_t WS_Q    = 36864;                      // f32 [E] scratch q
constexpr size_t WS_WV   = 65536;                      // bf16 [E][E]     2 MB
constexpr size_t WS_WOUT = WS_WV + 2u * kE * kE;       // bf16 [E][E]     2 MB
constexpr size_t WS_CTX  = WS_WOUT + 2u * kE * kE;     // bf16 [NTok][E] 32 MB

// -- CDNA5 async global->LDS copy (16B), tracked by ASYNCcnt ----------------
__device__ __forceinline__ void async_copy16(const float* g, uint32_t lds_off) {
  // GV mode: vdst = LDS byte address (low 32 bits of generic shared pointer),
  // vaddr = 64-bit global address.
  asm volatile("global_load_async_to_lds_b128 %0, %1, off"
               :: "v"(lds_off), "v"(g) : "memory");
}
__device__ __forceinline__ void wait_async0() {
  asm volatile("s_wait_asynccnt 0x0" ::: "memory");
}

// ---------------------------------------------------------------------------
// K0: q[f] = Wq[f,:] . query + bq[f]   (one wave per output row)
// ---------------------------------------------------------------------------
__global__ __launch_bounds__(256) void k_prep_q(
    const float* __restrict__ query, const float* __restrict__ in_w,
    const float* __restrict__ in_b, float* __restrict__ q_out) {
  const int lane = threadIdx.x & 31;
  const int wave = threadIdx.x >> 5;
  const int f = blockIdx.x * 8 + wave;           // grid 128 -> 1024 rows
  const float* wrow = in_w + (size_t)f * kE;
  float s = 0.f;
  for (int e = lane; e < kE; e += 32) s += wrow[e] * query[e];
  #pragma unroll
  for (int off = 16; off > 0; off >>= 1) s += __shfl_xor(s, off, 32);
  if (lane == 0) q_out[f] = s + in_b[f];
}

// ---------------------------------------------------------------------------
// K1: qk[h,e] = scale * sum_d q[h*D+d] * Wk[h*D+d, e];  qb[h] = scale * q_h.bk_h
// ---------------------------------------------------------------------------
__global__ __launch_bounds__(256) void k_prep_qk(
    const float* __restrict__ q, const float* __restrict__ in_w,
    const float* __restrict__ in_b, float* __restrict__ qk,
    float* __restrict__ qb) {
  __shared__ float qsh[kE];
  const int tid = threadIdx.x;
  for (int i = tid; i < kE; i += 256) qsh[i] = q[i];
  __syncthreads();
  const int i = blockIdx.x * 256 + tid;          // grid 32 -> 8192 outputs
  const int h = i >> 10, e = i & (kE - 1);
  float s = 0.f;
  const float* wk = in_w + (size_t)(kE + h * kD) * kE + e;  // Wk block
  #pragma unroll 4
  for (int d = 0; d < kD; ++d) s += qsh[h * kD + d] * wk[(size_t)d * kE];
  qk[i] = s * kScale;
  if (blockIdx.x == 0 && tid < kH) {
    float t = 0.f;
    for (int d = 0; d < kD; ++d) t += qsh[tid * kD + d] * in_b[kE + tid * kD + d];
    qb[tid] = t * kScale;
  }
}

// ---------------------------------------------------------------------------
// K2: convert Wv (rows 2E..3E of in_proj_weight) and Wout to bf16 in ws
// ---------------------------------------------------------------------------
__global__ __launch_bounds__(256) void k_convert(
    const float* __restrict__ in_w, const float* __restrict__ out_w,
    __bf16* __restrict__ wv_bf, __bf16* __restrict__ wout_bf) {
  const int i = blockIdx.x * 256 + threadIdx.x;  // grid 4096 -> 1M
  wv_bf[i]   = (__bf16)in_w[(size_t)2 * kE * kE + i];
  wout_bf[i] = (__bf16)out_w[i];
}

// ---------------------------------------------------------------------------
// K3: fused attention + context. One block = 16 tokens, 8 waves (1 wave/head).
//   Phase A: logits via qk dot x (LDS qk, shfl reductions), softmax,
//            writes `weights` output (mean over heads).
//   Phase B: per head h: ctx_tile(16 x 128) = Y(16 x 1024) @ Wv_h^T via
//            v_wmma_f32_16x16x32_bf16. x k-chunks are double-buffered into
//            LDS with global_load_async_to_lds_b128 (ASYNCcnt + barrier).
// ---------------------------------------------------------------------------
__global__ __launch_bounds__(256) void k_attn_ctx(
    const float* __restrict__ x,          // (S,B,L,E) fp32
    const float* __restrict__ in_b,       // (3E)
    const float* __restrict__ qk,         // ws [H][E] (pre-scaled)
    const float* __restrict__ qb,         // ws [H]    (pre-scaled)
    const __bf16* __restrict__ wv_bf,     // ws [E][E]
    __bf16* __restrict__ ctx_bf,          // ws [NTok][E]
    float* __restrict__ weights_out) {    // d_out tail, (S,B,L)
  __shared__ float qksh[kH][kE];          // 32 KB
  __shared__ float dotsh[16][kS][kH];     //  2 KB
  __shared__ float attnsh[16][kH][kS];    //  2 KB
  __shared__ float xs[2][16][kS][40];     // 20 KB, double-buffered, 16B-aligned rows

  const int tid  = threadIdx.x;
  const int lane = tid & 31;
  const int wave = tid >> 5;              // 0..7
  const int t0   = blockIdx.x * 16;
  const int b    = t0 / kL;
  const int l0   = t0 % kL;               // tile never straddles b (L%16==0)

  for (int i = tid; i < kH * kE; i += 256) qksh[i >> 10][i & (kE - 1)] = qk[i];
  __syncthreads();

  // ---- Phase A: logits ----------------------------------------------------
  for (int p = wave; p < 64; p += 8) {    // 64 (token, s) pairs
    const int m = p >> 2, s = p & 3;
    const float* xrow = x + (((size_t)(s * kB + b)) * kL + (l0 + m)) * kE;
    float sum[kH];
    #pragma unroll
    for (int h = 0; h < kH; ++h) sum[h] = 0.f;
    for (int e = lane; e < kE; e += 32) {
      const float xv = xrow[e];
      #pragma unroll
      for (int h = 0; h < kH; ++h) sum[h] += xv * qksh[h][e];
    }
    #pragma unroll
    for (int h = 0; h < kH; ++h) {
      #pragma unroll
      for (int off = 16; off > 0; off >>= 1) sum[h] += __shfl_xor(sum[h], off, 32);
    }
    if (lane == 0) {
      #pragma unroll
      for (int h = 0; h < kH; ++h) dotsh[m][s][h] = sum[h] + qb[h];
    }
  }
  __syncthreads();

  // ---- softmax over S=4 (one thread per (token, head)) --------------------
  if (tid < 128) {
    const int m = tid >> 3, h = tid & 7;
    const float d0 = dotsh[m][0][h], d1 = dotsh[m][1][h];
    const float d2 = dotsh[m][2][h], d3 = dotsh[m][3][h];
    const float mx = fmaxf(fmaxf(d0, d1), fmaxf(d2, d3));
    const float e0 = __expf(d0 - mx), e1 = __expf(d1 - mx);
    const float e2 = __expf(d2 - mx), e3 = __expf(d3 - mx);
    const float inv = 1.f / (e0 + e1 + e2 + e3);
    attnsh[m][h][0] = e0 * inv; attnsh[m][h][1] = e1 * inv;
    attnsh[m][h][2] = e2 * inv; attnsh[m][h][3] = e3 * inv;
  }
  __syncthreads();

  // ---- `weights` output: mean over heads, layout (S,B,L) ------------------
  if (tid < 64) {
    const int m = tid >> 2, s = tid & 3;
    float a = 0.f;
    #pragma unroll
    for (int h = 0; h < kH; ++h) a += attnsh[m][h][s];
    weights_out[((size_t)(s * kB + b)) * kL + (l0 + m)] = a * (1.f / kH);
  }

  // ---- Phase B: ctx via WMMA, async double-buffered x staging -------------
  const int h = wave;                     // one head per wave
  const int m = lane & 15;                // A-matrix row / B col / C col
  const int g = lane >> 4;                // lane group
  float am[kS];
  #pragma unroll
  for (int s = 0; s < kS; ++s) am[s] = attnsh[m][h][s];

  // per-thread staging slot: 8 consecutive floats of one (token, s) row
  const int smm = tid >> 4, srem = tid & 15;
  const int ss = srem >> 2, sq = srem & 3;
  const float* gbase =
      x + (((size_t)(ss * kB + b)) * kL + (l0 + smm)) * kE + sq * 8;
  const uint32_t ldst0 = (uint32_t)(uintptr_t)&xs[0][smm][ss][sq * 8];
  const uint32_t ldst1 = (uint32_t)(uintptr_t)&xs[1][smm][ss][sq * 8];

  // prologue: prefetch k-chunk 0 into buffer 0 (hits L2: phase A read it)
  async_copy16(gbase, ldst0);
  async_copy16(gbase + 4, ldst0 + 16);

  v8f acc[8];
  #pragma unroll
  for (int nt = 0; nt < 8; ++nt) { v8f z = {}; acc[nt] = z; }

  for (int kt = 0; kt < kE / 32; ++kt) {
    const int cur = kt & 1;
    wait_async0();                        // our loads for buf[cur] landed
    __syncthreads();                      // all waves landed + done w/ buf[1-cur]
    if (kt + 1 < kE / 32) {               // prefetch next chunk into buf[1-cur]
      const float* gnext = gbase + (kt + 1) * 32;
      const uint32_t dst = cur ? ldst0 : ldst1;
      async_copy16(gnext, dst);
      async_copy16(gnext + 4, dst + 16);
    }
    const int k0 = kt * 32;

    // Form bf16 A fragment: halves 0..7 <-> K=k0+16g+j, 8..15 <-> K=k0+16g+8+j
    float ylo[8], yhi[8];
    #pragma unroll
    for (int j = 0; j < 8; ++j) { ylo[j] = 0.f; yhi[j] = 0.f; }
    #pragma unroll
    for (int s = 0; s < kS; ++s) {
      const float* xr = &xs[cur][m][s][0];
      const float a = am[s];
      #pragma unroll
      for (int j = 0; j < 8; ++j) {
        ylo[j] += a * xr[16 * g + j];
        yhi[j] += a * xr[16 * g + 8 + j];
      }
    }
    v16bf afrag;
    #pragma unroll
    for (int j = 0; j < 8; ++j) {
      afrag[j]     = (__bf16)ylo[j];
      afrag[j + 8] = (__bf16)yhi[j];
    }

    #pragma unroll
    for (int nt = 0; nt < 8; ++nt) {      // 8 n-tiles cover D=128 for head h
      const __bf16* brow =
          wv_bf + ((size_t)(h * kD + nt * 16 + m)) * kE + k0 + 16 * g;
      const v16bf bfrag = *(const v16bf*)brow;
      acc[nt] = __builtin_amdgcn_wmma_f32_16x16x32_bf16(
          false, afrag, false, bfrag, (short)0, acc[nt], false, false);
    }
  }

  // epilogue: +bv (softmax sums to 1), bf16 ctx for the out-proj GEMM
  #pragma unroll
  for (int nt = 0; nt < 8; ++nt) {
    const int col = h * kD + nt * 16 + m;
    const float bv = in_b[2 * kE + col];
    #pragma unroll
    for (int r = 0; r < 8; ++r) {
      const int trow = t0 + r + 8 * g;    // C layout: M = r + 8*laneGroup
      ctx_bf[(size_t)trow * kE + col] = (__bf16)(acc[nt][r] + bv);
    }
  }
}

// ---------------------------------------------------------------------------
// K4: aggregated = ctx @ Wout^T + bout. Wave-tiled bf16 WMMA GEMM,
//     M=16384, N=1024, K=1024. Block = 8 waves -> 16x128 output tile.
// ---------------------------------------------------------------------------
__global__ __launch_bounds__(256) void k_outproj(
    const __bf16* __restrict__ ctx_bf, const __bf16* __restrict__ wout_bf,
    const float* __restrict__ out_b, float* __restrict__ out) {
  const int lane = threadIdx.x & 31;
  const int wave = threadIdx.x >> 5;
  const int t0 = blockIdx.x * 16;
  const int n0 = blockIdx.y * 128 + wave * 16;
  const int m = lane & 15, g = lane >> 4;

  v8f acc = {};
  for (int k0 = 0; k0 < kE; k0 += 32) {
    const __bf16* arow = ctx_bf + (size_t)(t0 + m) * kE + k0 + 16 * g;
    const v8bf alo = *(const v8bf*)arow;
    const v8bf ahi = *(const v8bf*)(arow + 8);
    const v16bf afrag = __builtin_shufflevector(
        alo, ahi, 0, 1, 2, 3, 4, 5, 6, 7, 8, 9, 10, 11, 12, 13, 14, 15);
    const __bf16* brow = wout_bf + (size_t)(n0 + m) * kE + k0 + 16 * g;
    const v16bf bfrag = *(const v16bf*)brow;
    acc = __builtin_amdgcn_wmma_f32_16x16x32_bf16(
        false, afrag, false, bfrag, (short)0, acc, false, false);
  }
  const int n = n0 + m;
  const float bias = out_b[n];
  #pragma unroll
  for (int r = 0; r < 8; ++r)
    out[(size_t)(t0 + r + 8 * g) * kE + n] = acc[r] + bias;
}

// ---------------------------------------------------------------------------
extern "C" void kernel_launch(void* const* d_in, const int* in_sizes, int n_in,
                              void* d_out, int out_size, void* d_ws,
                              size_t ws_size, hipStream_t stream) {
  const float* x      = (const float*)d_in[0];  // stream_outputs (S,B,L,E)
  const float* query  = (const float*)d_in[1];  // (1,1,E)
  const float* in_w   = (const float*)d_in[2];  // (3E,E)
  const float* in_b   = (const float*)d_in[3];  // (3E)
  const float* out_w  = (const float*)d_in[4];  // (E,E)
  const float* out_b  = (const float*)d_in[5];  // (E)
  float* out = (float*)d_out;                   // agg (B,L,E) then weights (S,B,L)

  char* ws = (char*)d_ws;                       // needs ~38 MB
  float*  qk      = (float*)(ws + WS_QK);
  float*  qb      = (float*)(ws + WS_QB);
  float*  q_tmp   = (float*)(ws + WS_Q);
  __bf16* wv_bf   = (__bf16*)(ws + WS_WV);
  __bf16* wout_bf = (__bf16*)(ws + WS_WOUT);
  __bf16* ctx_bf  = (__bf16*)(ws + WS_CTX);

  k_prep_q<<<kE / 8, 256, 0, stream>>>(query, in_w, in_b, q_tmp);
  k_prep_qk<<<(kH * kE) / 256, 256, 0, stream>>>(q_tmp, in_w, in_b, qk, qb);
  k_convert<<<(kE * kE) / 256, 256, 0, stream>>>(in_w, out_w, wv_bf, wout_bf);
  k_attn_ctx<<<kNTok / 16, 256, 0, stream>>>(x, in_b, qk, qb, wv_bf, ctx_bf,
                                             out + (size_t)kNTok * kE);
  dim3 g4(kNTok / 16, kE / 128);
  k_outproj<<<g4, 256, 0, stream>>>(ctx_bf, wout_bf, out_b, out);
}
